// SimpleRNNCell_64467459113108
// MI455X (gfx1250) — matrix operand order repaired and checked by reading, not compiled, not run on hardware
//
#include <hip/hip_runtime.h>

// ---------------------------------------------------------------------------
// SimpleRNN linear recurrence on gfx1250 (MI455X), wave32 + WMMA bf16.
//   Phase 1: xw = x @ W  (16384 x 1024 x 1024) -> d_out
//            B staged through LDS with global_load_async_to_lds_b128
//            (double-buffered, s_wait_asynccnt + barriers)
//   Phase 2: h_t = xw_t + h_{t-1} @ U, 512 sequential step kernels
// bf16 operands, fp32 accumulation via v_wmma_f32_16x16x32_bf16.
// ---------------------------------------------------------------------------

typedef __attribute__((ext_vector_type(16))) __bf16 v16bf;
typedef __attribute__((ext_vector_type(8)))  float  v8f;

#define RNN_B 32
#define RNN_T 512
#define RNN_D 1024   // K == N == D

// B-strip LDS row stride: 32 bf16 data + 8 bf16 pad = 80 bytes (16B-aligned,
// spreads DS banks across the 16 lanes reading consecutive rows).
#define BROW 40

__device__ __forceinline__ unsigned short f32_to_bf16(float f) {
    unsigned u = __builtin_bit_cast(unsigned, f);
    u += 0x7fffu + ((u >> 16) & 1u);          // round-to-nearest-even
    return (unsigned short)(u >> 16);
}

// D = A(16x32 bf16) * B(32x16 bf16) + C(16x16 f32)
__device__ __forceinline__ v8f wmma_bf16(v16bf a, v16bf b, v8f c) {
    return __builtin_amdgcn_wmma_f32_16x16x32_bf16(
        /*neg_a=*/false, a, /*neg_b=*/false, b,
        /*c_mod=*/(short)0, c, /*reuse_a=*/false, /*reuse_b=*/false);
}

// Async copy of 16 bytes global -> LDS (per lane). Tracked by ASYNCcnt.
__device__ __forceinline__ void async_b128(const unsigned short* gsrc,
                                           const unsigned short* ldst) {
    // Low 32 bits of the generic shared-aperture address == LDS byte offset.
    unsigned loff = (unsigned)(size_t)ldst;
    asm volatile("global_load_async_to_lds_b128 %0, %1, off"
                 :: "v"(loff), "v"(gsrc)
                 : "memory");
}

#define WAIT_ASYNC(n) asm volatile("s_wait_asynccnt %0" :: "i"(n) : "memory")

// Build the 16x32 bf16 A fragment for this lane from an fp32 row.
// Contiguous K order used here, same as B (16-bit A 16x32): lane<16 -> M=lane, K=k0..k0+15
//                              lane>=16 -> M=lane-16, K=k0+16..k0+31.
__device__ __forceinline__ v16bf make_a(const float4* __restrict__ row4,
                                        int k0, int hi) {
    const int base = (k0 >> 2) + (hi << 2);   // (k0 + hi*16)/4
    float4 f0 = row4[base];
    float4 f1 = row4[base + 1];
    float4 f2 = row4[base + 2];               // (k0 + 8 + hi*16)/4
    float4 f3 = row4[base + 3];
    v16bf v;
    v[0]=(__bf16)f0.x;  v[1]=(__bf16)f0.y;  v[2]=(__bf16)f0.z;  v[3]=(__bf16)f0.w;
    v[4]=(__bf16)f1.x;  v[5]=(__bf16)f1.y;  v[6]=(__bf16)f1.z;  v[7]=(__bf16)f1.w;
    v[8]=(__bf16)f2.x;  v[9]=(__bf16)f2.y;  v[10]=(__bf16)f2.z; v[11]=(__bf16)f2.w;
    v[12]=(__bf16)f3.x; v[13]=(__bf16)f3.y; v[14]=(__bf16)f3.z; v[15]=(__bf16)f3.w;
    return v;
}

// Build the 32x16 bf16 B fragment from memory in [N][K] order.
// Contiguous K order used here, same as A (16-bit B 32x16): lanes 0-15 hold K=k0..k0+15 (N=lane),
//                              lanes 16-31 hold K=k0+16..k0+31 (N=lane-16).
__device__ __forceinline__ v16bf make_b16(const unsigned short* __restrict__ p) {
    union { uint4 q[2]; v16bf v; } u;
    u.q[0] = *(const uint4*)(p);
    u.q[1] = *(const uint4*)(p + 8);
    return u.v;
}

// ---------------------------------------------------------------------------
// Kernel 0: transpose-convert W and U (fp32 [K][N]) -> bf16 [N][K]
// ---------------------------------------------------------------------------
__global__ __launch_bounds__(256)
void cvt_transpose_kernel(const float* __restrict__ W,
                          const float* __restrict__ U,
                          unsigned short* __restrict__ Wt,
                          unsigned short* __restrict__ Ut) {
    __shared__ float tile[32][33];
    const float*    src = blockIdx.z ? U  : W;
    unsigned short* dst = blockIdx.z ? Ut : Wt;
    const int bx = blockIdx.x * 32;   // n (column of src)
    const int by = blockIdx.y * 32;   // k (row of src)
    const int tx = threadIdx.x, ty = threadIdx.y;   // 32 x 8
#pragma unroll
    for (int i = 0; i < 32; i += 8)
        tile[ty + i][tx] = src[(size_t)(by + ty + i) * RNN_D + (bx + tx)];
    __syncthreads();
#pragma unroll
    for (int i = 0; i < 32; i += 8)   // dst[n][k] = src[k][n]
        dst[(size_t)(bx + ty + i) * RNN_D + (by + tx)] = f32_to_bf16(tile[tx][ty + i]);
}

// ---------------------------------------------------------------------------
// Kernel 1: xw = x @ W.
// Block = 128 threads = 4 waves; block tile = 128 rows x 64 cols.
// Wave w owns M-tiles {2w, 2w+1} (32 rows) x 4 N-tiles -> 8 accumulators.
// The shared 32(K) x 64(N) B strip is async-copied to LDS, double-buffered.
// ---------------------------------------------------------------------------
__device__ __forceinline__ void gemm_chunk(const float4* __restrict__ arow0,
                                           const float4* __restrict__ arow1,
                                           const unsigned short* __restrict__ bbuf,
                                           int k0, int hi, int l15,
                                           v8f acc[2][4]) {
    v16bf a0 = make_a(arow0, k0, hi);
    v16bf a1 = make_a(arow1, k0, hi);
#pragma unroll
    for (int j = 0; j < 4; ++j) {
        v16bf b = make_b16(bbuf + (j * 16 + l15) * BROW + hi * 16);  // ds_load_b128 x2
        acc[0][j] = wmma_bf16(a0, b, acc[0][j]);
        acc[1][j] = wmma_bf16(a1, b, acc[1][j]);
    }
}

__global__ __launch_bounds__(128)
void gemm_xw_kernel(const float* __restrict__ x,           // [16384][1024] fp32
                    const unsigned short* __restrict__ Wt, // [1024][1024] bf16 (N-major)
                    float* __restrict__ out) {             // [16384][1024] fp32
    __shared__ __align__(16) unsigned short bsm[2][64 * BROW];

    const int lane = threadIdx.x & 31;
    const int wave = threadIdx.x >> 5;
    const int l15  = lane & 15;
    const int hi   = lane >> 4;
    const int mbase = blockIdx.x * 128 + wave * 32;   // first of 32 rows this wave
    const int nbase = blockIdx.y * 64;

    const float4* arow0 = (const float4*)(x + (size_t)(mbase + l15) * RNN_D);
    const float4* arow1 = (const float4*)(x + (size_t)(mbase + 16 + l15) * RNN_D);

    // Issue the async copy of one 32(K) x 64(N) bf16 strip (4 KB): 256 x 16B
    // segments spread over 128 threads x 2.
    auto issue_chunk = [&](int k0, int buf) {
        const int t = threadIdx.x;
#pragma unroll
        for (int s = 0; s < 2; ++s) {
            const int seg = t + s * 128;
            const int row = seg >> 2;       // 0..63 (N within strip)
            const int q   = seg & 3;        // 16B quarter of the 64B row
            async_b128(Wt + (size_t)(nbase + row) * RNN_D + k0 + q * 8,
                       &bsm[buf][row * BROW + q * 8]);
        }
    };

    v8f acc[2][4] = {};

    issue_chunk(0, 0);
    for (int k0 = 0; k0 < RNN_D - 32; k0 += 32) {
        const int cur = (k0 >> 5) & 1;
        __syncthreads();                       // buf cur^1 free for reuse
        issue_chunk(k0 + 32, cur ^ 1);         // prefetch next strip
        // Branchless A prefetch two chunks ahead (clamped to stay in-bounds).
        const int kp = (k0 + 64 < RNN_D) ? (k0 + 64) : k0;
        __builtin_prefetch(arow0 + (kp >> 2), 0, 3);
        WAIT_ASYNC(2);                         // strip k0 landed (in-order)
        __syncthreads();                       // ... for every wave
        gemm_chunk(arow0, arow1, bsm[cur], k0, hi, l15, acc);
    }
    __syncthreads();
    WAIT_ASYNC(0);
    __syncthreads();
    gemm_chunk(arow0, arow1, bsm[1], RNN_D - 32, hi, l15, acc);

    // C/D layout: lane<16 -> (M=r, N=lane); lane>=16 -> (M=8+r, N=lane-16)
#pragma unroll
    for (int mi = 0; mi < 2; ++mi)
#pragma unroll
        for (int j = 0; j < 4; ++j) {
            const int col = nbase + j * 16 + l15;
#pragma unroll
            for (int r = 0; r < 8; ++r) {
                const int row = mbase + mi * 16 + hi * 8 + r;
                out[(size_t)row * RNN_D + col] = acc[mi][j][r];
            }
        }
}

// ---------------------------------------------------------------------------
// Kernel 2 (x512, sequential): h_t = xw_t + h_{t-1} @ U.
// Block = 256 threads = 8 waves; wave owns a 32x32 block (2x2 tiles).
// Grid.x = 1024/(8*32) = 4.  outT points at d_out[:, t, :] (row stride T*D).
// ---------------------------------------------------------------------------
__global__ __launch_bounds__(256)
void rnn_step_kernel(const float* __restrict__ hprev,       // [32][1024] fp32
                     const unsigned short* __restrict__ Ut, // [1024][1024] bf16 (N-major)
                     float* __restrict__ outT,              // d_out + t*D
                     float* __restrict__ hnext) {           // [32][1024] fp32
    const size_t rstride = (size_t)RNN_T * RNN_D;           // batch row stride in d_out
    const int lane = threadIdx.x & 31;
    const int wave = threadIdx.x >> 5;
    const int l15  = lane & 15;
    const int hi   = lane >> 4;
    const int ncol = (blockIdx.x * 8 + wave) * 32;

    const float4* arow0 = (const float4*)(hprev + (size_t)l15 * RNN_D);
    const float4* arow1 = (const float4*)(hprev + (size_t)(16 + l15) * RNN_D);
    const unsigned short* brow0 = Ut + (size_t)(ncol + l15) * RNN_D;
    const unsigned short* brow1 = Ut + (size_t)(ncol + 16 + l15) * RNN_D;
    v8f acc[2][2] = {};

#pragma unroll 2
    for (int k0 = 0; k0 < RNN_D; k0 += 32) {
        v16bf a0 = make_a(arow0, k0, hi);
        v16bf a1 = make_a(arow1, k0, hi);
        v16bf b0 = make_b16(brow0 + k0 + hi * 16);
        v16bf b1 = make_b16(brow1 + k0 + hi * 16);
        acc[0][0] = wmma_bf16(a0, b0, acc[0][0]);
        acc[0][1] = wmma_bf16(a0, b1, acc[0][1]);
        acc[1][0] = wmma_bf16(a1, b0, acc[1][0]);
        acc[1][1] = wmma_bf16(a1, b1, acc[1][1]);
    }

#pragma unroll
    for (int mt = 0; mt < 2; ++mt)
#pragma unroll
        for (int nt = 0; nt < 2; ++nt) {
            const int col = ncol + nt * 16 + l15;
#pragma unroll
            for (int r = 0; r < 8; ++r) {
                const int row = mt * 16 + hi * 8 + r;
                const float v = acc[mt][nt][r] + outT[(size_t)row * rstride + col];
                outT[(size_t)row * rstride + col] = v;          // h_t -> output slice
                hnext[row * RNN_D + col]          = v;          // h_t -> carry buffer
            }
        }
}

// ---------------------------------------------------------------------------
extern "C" void kernel_launch(void* const* d_in, const int* in_sizes, int n_in,
                              void* d_out, int out_size, void* d_ws, size_t ws_size,
                              hipStream_t stream) {
    const float* x  = (const float*)d_in[0];  // [B,T,D]
    const float* W  = (const float*)d_in[1];  // [D,D]
    const float* U  = (const float*)d_in[2];  // [D,D]
    const float* h0 = (const float*)d_in[3];  // [B,D]
    float* out = (float*)d_out;               // [B,T,D]

    // Workspace layout: Wt (2MB) | Ut (2MB) | hbuf0 (128KB) | hbuf1 (128KB)
    unsigned short* Wt = (unsigned short*)d_ws;
    unsigned short* Ut = Wt + (size_t)RNN_D * RNN_D;
    float* hb0 = (float*)(Ut + (size_t)RNN_D * RNN_D);
    float* hb1 = hb0 + (size_t)RNN_B * RNN_D;

    // 0) weights -> bf16, transposed to [N][K]
    cvt_transpose_kernel<<<dim3(RNN_D / 32, RNN_D / 32, 2), dim3(32, 8), 0, stream>>>(
        W, U, Wt, Ut);

    // 1) xw = x @ W  (M = B*T); block tile 128x64
    gemm_xw_kernel<<<dim3((RNN_B * RNN_T) / 128, RNN_D / 64), 128, 0, stream>>>(
        x, Wt, out);

    // 2) sequential scan: h_t = xw_t + h_{t-1} @ U, ping-pong h buffers
    for (int t = 0; t < RNN_T; ++t) {
        const float* hsrc = (t == 0) ? h0 : ((t & 1) ? hb0 : hb1);
        float*       hdst = (t & 1) ? hb1 : hb0;
        rnn_step_kernel<<<dim3(RNN_D / 256), 256, 0, stream>>>(
            hsrc, Ut, out + (size_t)t * RNN_D, hdst);
    }
}
